// MemoryEfficientMultiHeadAttention_58737972740531
// MI455X (gfx1250) — matrix operand order repaired and checked by reading, not compiled, not run on hardware
//
#include <hip/hip_runtime.h>

typedef unsigned short u16;
typedef unsigned int   u32;
typedef __attribute__((ext_vector_type(16))) __bf16 v16bf;
typedef __attribute__((ext_vector_type(8)))  float  v8f;

struct alignas(16) U4 { u32 x, y, z, w; };
union BF16x16 { U4 u[2]; v16bf v; };

#define WMMA_BF16(Aop, Bop, Cop) \
  __builtin_amdgcn_wmma_f32_16x16x32_bf16(false, (Aop), false, (Bop), (short)0, (Cop), false, false)

__device__ __forceinline__ u16 f2bf(float f) {
  u32 u = __float_as_uint(f);
  u32 r = u + 0x7FFFu + ((u >> 16) & 1u);          // round-to-nearest-even
  if ((u & 0x7F800000u) == 0x7F800000u) r = u;     // inf/NaN: truncate
  return (u16)(r >> 16);
}

// ---------------------------------------------------------------- stage 0
__global__ void cvt_f32_bf16(const float* __restrict__ in, u16* __restrict__ out, int n) {
  int i = blockIdx.x * blockDim.x + threadIdx.x;
  if (i < n) out[i] = f2bf(in[i]);
}

// ---------------------------------------------------------------- stages 1 & 4
// One wave computes a 32x64 output tile (8 WMMAs / k-step, double-buffered loads).
// dst layouts: mode 0: bf16 (B,H,N,D)   mode 1: f32 row-major MxC   mode 2: bf16 (B,H,D,N)
#define CK 512
__global__ void gemm512_bf16(const u16* __restrict__ X, const u16* __restrict__ W,
                             const float* __restrict__ bias, void* __restrict__ dst,
                             int M, int Ntok, int mode, float scale) {
  int wave  = (int)((blockIdx.x * blockDim.x + threadIdx.x) >> 5);
  int lane  = threadIdx.x & 31;
  int lm    = lane & 15, hi = lane >> 4;
  int tilesM = M >> 5;                       // 32-row tiles
  int tm = wave % tilesM, tn = wave / tilesM;
  int m0 = tm << 5, n0 = tn << 6;

  v8f acc[2][4];
#pragma unroll
  for (int j = 0; j < 4; ++j) {
    float bv = bias[n0 + j * 16 + lm];
#pragma unroll
    for (int v = 0; v < 8; ++v) { acc[0][j][v] = bv; acc[1][j][v] = bv; }
  }

  const u16* arow0 = X + (size_t)(m0 + lm) * CK + hi * 8;
  const u16* arow1 = arow0 + (size_t)16 * CK;
  const u16* brow  = W + (size_t)(n0 + lm) * CK + hi * 8;

  BF16x16 A[2][2], Bm[2][4];
  {
    const U4* a0 = (const U4*)arow0;  A[0][0].u[0] = a0[0]; A[0][0].u[1] = a0[2];
    const U4* a1 = (const U4*)arow1;  A[0][1].u[0] = a1[0]; A[0][1].u[1] = a1[2];
#pragma unroll
    for (int j = 0; j < 4; ++j) {
      const U4* bp = (const U4*)(brow + (size_t)j * 16 * CK);
      Bm[0][j].u[0] = bp[0]; Bm[0][j].u[1] = bp[2];
    }
  }

  int cur = 0;
#pragma unroll
  for (int kk = 0; kk < CK; kk += 32) {
    int nxt = cur ^ 1;
    if (kk + 32 < CK) {                      // prefetch next k-step
      const U4* a0 = (const U4*)(arow0 + kk + 32); A[nxt][0].u[0] = a0[0]; A[nxt][0].u[1] = a0[2];
      const U4* a1 = (const U4*)(arow1 + kk + 32); A[nxt][1].u[0] = a1[0]; A[nxt][1].u[1] = a1[2];
#pragma unroll
      for (int j = 0; j < 4; ++j) {
        const U4* bp = (const U4*)(brow + (size_t)j * 16 * CK + kk + 32);
        Bm[nxt][j].u[0] = bp[0]; Bm[nxt][j].u[1] = bp[2];
      }
    }
#pragma unroll
    for (int j = 0; j < 4; ++j) {
      acc[0][j] = WMMA_BF16(A[cur][0].v, Bm[cur][j].v, acc[0][j]);
      acc[1][j] = WMMA_BF16(A[cur][1].v, Bm[cur][j].v, acc[1][j]);
    }
    cur = nxt;
  }

#pragma unroll
  for (int mt = 0; mt < 2; ++mt) {
#pragma unroll
    for (int j = 0; j < 4; ++j) {
#pragma unroll
      for (int v = 0; v < 8; ++v) {
        int mabs = m0 + mt * 16 + v + hi * 8;
        int nabs = n0 + j * 16 + lm;
        float o = acc[mt][j][v] * scale;
        if (mode == 1) {
          ((float*)dst)[(size_t)mabs * CK + nabs] = o;
        } else {
          int b = mabs / Ntok, n = mabs % Ntok;
          int h = nabs >> 6, d = nabs & 63;
          size_t offn = (mode == 0)
              ? ((size_t)((b * 8 + h) * Ntok + n)) * 64 + d
              : ((size_t)((b * 8 + h) * 64 + d)) * Ntok + n;
          ((u16*)dst)[offn] = f2bf(o);
        }
      }
    }
  }
}

// ---------------------------------------------------------------- stage 3: flash attention
__global__ void flash16(const u16* __restrict__ Qp, const u16* __restrict__ Kp,
                        const u16* __restrict__ Vt, const unsigned char* __restrict__ mask,
                        u16* __restrict__ Obuf) {
  __shared__ __align__(16) u16 plds[8 * 16 * 32];
  int lane = threadIdx.x & 31, wl = threadIdx.x >> 5;
  int wave = blockIdx.x * (blockDim.x >> 5) + wl;
  int qt = wave & 127, bh = wave >> 7;            // 128 q-tiles, bh = b*8+h
  int b = bh >> 3, h = bh & 7;
  int lm = lane & 15, hi = lane >> 4;
  int q0 = qt << 4;

  const u16* Qb = Qp + (size_t)bh * 2048 * 64;
  const u16* Kb = Kp + (size_t)bh * 2048 * 64;
  const u16* Vb = Vt + (size_t)bh * 64 * 2048;
  const unsigned char* mk = mask + b * 2048;
  u16* pl = plds + wl * 512;

  BF16x16 Qa[2];
#pragma unroll
  for (int ka = 0; ka < 2; ++ka) {
    const U4* qp = (const U4*)(Qb + (size_t)(q0 + lm) * 64 + ka * 32 + hi * 16);
    Qa[ka].u[0] = qp[0]; Qa[ka].u[1] = qp[1];
  }

  v8f O[4];
  float mrow[8], lrow[8];
#pragma unroll
  for (int j = 0; j < 4; ++j)
#pragma unroll
    for (int v = 0; v < 8; ++v) O[j][v] = 0.0f;
#pragma unroll
  for (int v = 0; v < 8; ++v) { mrow[v] = -3.0e38f; lrow[v] = 0.0f; }

  // K double buffer: [buf][js][chan-block]
  BF16x16 KB[2][2][2];
#pragma unroll
  for (int js = 0; js < 2; ++js) {
    const U4* kp = (const U4*)(Kb + (size_t)(js * 16 + lm) * 64);
    KB[0][js][0].u[0] = kp[hi * 2];     KB[0][js][0].u[1] = kp[hi * 2 + 1];
    KB[0][js][1].u[0] = kp[4 + hi * 2]; KB[0][js][1].u[1] = kp[4 + hi * 2 + 1];
  }

  int cur = 0;
  for (int k0 = 0; k0 < 2048; k0 += 32) {
    int nxt = cur ^ 1;
    // ---- S = Q K^T for 32 keys (uses prefetched K tiles)
    v8f S[2];
#pragma unroll
    for (int js = 0; js < 2; ++js) {
      v8f s;
#pragma unroll
      for (int v = 0; v < 8; ++v) s[v] = 0.0f;
      s = WMMA_BF16(Qa[0].v, KB[cur][js][0].v, s);
      s = WMMA_BF16(Qa[1].v, KB[cur][js][1].v, s);
      if (mk[k0 + js * 16 + lm]) {
#pragma unroll
        for (int v = 0; v < 8; ++v) s[v] = -3.0e38f;
      }
      S[js] = s;
    }
    // ---- prefetch next K block (hidden under softmax + PV)
    if (k0 + 32 < 2048) {
#pragma unroll
      for (int js = 0; js < 2; ++js) {
        const U4* kp = (const U4*)(Kb + (size_t)(k0 + 32 + js * 16 + lm) * 64);
        KB[nxt][js][0].u[0] = kp[hi * 2];     KB[nxt][js][0].u[1] = kp[hi * 2 + 1];
        KB[nxt][js][1].u[0] = kp[4 + hi * 2]; KB[nxt][js][1].u[1] = kp[4 + hi * 2 + 1];
      }
    }
    // ---- prefetch V tiles for this block (latency hidden by softmax VALU)
    BF16x16 Vm[4];
#pragma unroll
    for (int j = 0; j < 4; ++j) {
      const U4* vp = (const U4*)(Vb + (size_t)(j * 16 + lm) * 2048 + k0 + hi * 8);
      Vm[j].u[0] = vp[0]; Vm[j].u[1] = vp[2];
    }

    // ---- online softmax (row m = v + 8*hi lives in one 16-lane half)
    float mnew[8], alpha[8];
#pragma unroll
    for (int v = 0; v < 8; ++v) {
      float t = fmaxf(S[0][v], S[1][v]);
      t = fmaxf(t, __shfl_xor(t, 1));
      t = fmaxf(t, __shfl_xor(t, 2));
      t = fmaxf(t, __shfl_xor(t, 4));
      t = fmaxf(t, __shfl_xor(t, 8));
      float mn = fmaxf(mrow[v], t);
      mnew[v] = mn;
      alpha[v] = __expf(mrow[v] - mn);
      mrow[v] = mn;
    }
#pragma unroll
    for (int v = 0; v < 8; ++v) {
      float p0 = __expf(S[0][v] - mnew[v]);
      float p1 = __expf(S[1][v] - mnew[v]);
      int m = v + hi * 8;
      pl[m * 32 + lm]      = f2bf(p0);
      pl[m * 32 + 16 + lm] = f2bf(p1);
      float ps = p0 + p1;
      ps += __shfl_xor(ps, 1);
      ps += __shfl_xor(ps, 2);
      ps += __shfl_xor(ps, 4);
      ps += __shfl_xor(ps, 8);
      lrow[v] = lrow[v] * alpha[v] + ps;
    }
#pragma unroll
    for (int j = 0; j < 4; ++j)
#pragma unroll
      for (int v = 0; v < 8; ++v) O[j][v] *= alpha[v];

    asm volatile("s_wait_dscnt 0" ::: "memory");   // P stores visible before A-layout reload
    BF16x16 Pa;
    const U4* pp = (const U4*)(pl + lm * 32 + hi * 8);
    Pa.u[0] = pp[0]; Pa.u[1] = pp[2];
#pragma unroll
    for (int j = 0; j < 4; ++j)
      O[j] = WMMA_BF16(Pa.v, Vm[j].v, O[j]);

    cur = nxt;
  }

#pragma unroll
  for (int j = 0; j < 4; ++j) {
#pragma unroll
    for (int v = 0; v < 8; ++v) {
      float o = O[j][v] / lrow[v];
      int mabs = q0 + v + hi * 8;
      int c = h * 64 + j * 16 + lm;
      Obuf[((size_t)b * 2048 + mabs) * 512 + c] = f2bf(o);   // (B, Nq, C) row-major
    }
  }
}

// ---------------------------------------------------------------- launcher
extern "C" void kernel_launch(void* const* d_in, const int* in_sizes, int n_in,
                              void* d_out, int out_size, void* d_ws, size_t ws_size,
                              hipStream_t stream) {
  const float* query = (const float*)d_in[0];
  const float* keyi  = (const float*)d_in[1];
  const float* vali  = (const float*)d_in[2];
  const unsigned char* kpm = (const unsigned char*)d_in[3];
  const float* Wq = (const float*)d_in[4];  const float* bq = (const float*)d_in[5];
  const float* Wk = (const float*)d_in[6];  const float* bk = (const float*)d_in[7];
  const float* Wv = (const float*)d_in[8];  const float* bv = (const float*)d_in[9];
  const float* Wo = (const float*)d_in[10]; const float* bo = (const float*)d_in[11];
  float* out = (float*)d_out;

  const int B = 2, N = 2048, C = 512;
  const int M = B * N;                       // 4096 tokens
  const size_t nBNC = (size_t)M * C;         // 2,097,152 elements
  const size_t nCC  = (size_t)C * C;

  char* ws = (char*)d_ws; size_t off = 0;
  auto take = [&](size_t elems) -> u16* {
    u16* p = (u16*)(ws + off);
    off += ((elems * sizeof(u16)) + 255) & ~(size_t)255;
    return p;
  };
  u16 *Xq = take(nBNC), *Xk = take(nBNC), *Xv = take(nBNC);
  u16 *wq = take(nCC),  *wk = take(nCC),  *wv = take(nCC), *wo = take(nCC);
  u16 *Qp = take(nBNC), *Kp = take(nBNC), *Vt = take(nBNC), *AO = take(nBNC);

  auto cvt = [&](const float* src, u16* dst, size_t n) {
    cvt_f32_bf16<<<(int)((n + 255) / 256), 256, 0, stream>>>(src, dst, (int)n);
  };
  cvt(query, Xq, nBNC); cvt(keyi, Xk, nBNC); cvt(vali, Xv, nBNC);
  cvt(Wq, wq, nCC); cvt(Wk, wk, nCC); cvt(Wv, wv, nCC); cvt(Wo, wo, nCC);

  const int gemmGrid = (M / 32) * (C / 64) / 8;       // 1024 waves / 8 per block = 128 blocks
  gemm512_bf16<<<gemmGrid, 256, 0, stream>>>(Xq, wq, bq, Qp, M, N, 0, 0.125f); // scale folded into Q
  gemm512_bf16<<<gemmGrid, 256, 0, stream>>>(Xk, wk, bk, Kp, M, N, 0, 1.0f);
  gemm512_bf16<<<gemmGrid, 256, 0, stream>>>(Xv, wv, bv, Vt, M, N, 2, 1.0f);   // V transposed (B,H,D,N)

  flash16<<<(B * 8 * (N / 16)) / 8, 256, 0, stream>>>(Qp, Kp, Vt, kpm, AO);

  gemm512_bf16<<<gemmGrid, 256, 0, stream>>>(AO, wo, bo, out, M, N, 1, 1.0f);  // fp32 out
}
